// ExtractorHead_18451179503857
// MI455X (gfx1250) — matrix operand order repaired and checked by reading, not compiled, not run on hardware
//
#include <hip/hip_runtime.h>
#include <hip/hip_bf16.h>

#define H 128
#define H3 384
#define NRBF 20
#define NLAYER 3
#define PI_F 3.14159265358979f
#define CUT_F 6.0f

typedef __attribute__((ext_vector_type(16))) __bf16        bf16x16;
typedef __attribute__((ext_vector_type(16))) short         short16;
typedef __attribute__((ext_vector_type(8)))  float         f32x8;
typedef __attribute__((ext_vector_type(8)))  unsigned int  uint32x8;

// ---------- helpers ----------
__device__ __forceinline__ unsigned short f2bf(float f) {
  unsigned u = __float_as_uint(f);
  u += 0x7fffu + ((u >> 16) & 1u);           // round-to-nearest-even
  return (unsigned short)(u >> 16);
}

__device__ __forceinline__ float silu_f(float x) {
  // x * rcp(1+exp(-x)) : v_exp_f32 + v_rcp_f32, no IEEE-div sequence
  return x * __builtin_amdgcn_rcpf(1.0f + __expf(-x));
}

__device__ __forceinline__ f32x8 wmma_bf16(short16 a, short16 b, f32x8 c) {
  return __builtin_amdgcn_wmma_f32_16x16x32_bf16(
      false, __builtin_bit_cast(bf16x16, a),
      false, __builtin_bit_cast(bf16x16, b),
      (short)0, c, false, false);
}

// Packed fragment layouts (ISA 7.12.2): one uint32 = bf16 pair for one VGPR
// slot of one lane. A lane's 8 dwords are contiguous -> 32B vector load.
// A (16x32): lane = lanehalf*16 + m ; koff 0-7 -> half0 p0-3, 8-15 -> half1
// p0-3, 16-23 -> half0 p4-7, 24-31 -> half1 p4-7.
// B (32x16): lane = ((k>>3)&1)*16 + n ; same k order as A (p holds k = 8h+16(p>>2)+2(p&3) and +1).
__device__ __forceinline__ short16 load_a_packed(const unsigned int* As, int kb) {
  const int lane = threadIdx.x & 31;
  const uint32x8 v = *(const uint32x8*)(As + (size_t)((kb << 5) + lane) * 8);
  return __builtin_bit_cast(short16, v);
}

__device__ __forceinline__ short16 load_b_packed(const unsigned int* P, int ntn, int kt, int nt) {
  const int lane = threadIdx.x & 31;
  const uint32x8 v = *(const uint32x8*)(P + ((size_t)(kt * ntn + nt) * 32 + lane) * 8);
  return __builtin_bit_cast(short16, v);
}

// ---------- weight repack: row-major f32 [K][Nc] -> packed bf16 fragments ----------
__global__ void k_pack_w(const float* __restrict__ W, unsigned int* __restrict__ P,
                         int K, int Nc) {
  int q = blockIdx.x * blockDim.x + threadIdx.x;
  int total = (K >> 1) * Nc;
  if (q >= total) return;
  int p    = q & 7;
  int lane = (q >> 3) & 31;
  int tile = q >> 8;
  int ntn  = Nc >> 4;
  int nt   = tile % ntn;
  int kt   = tile / ntn;
  int n  = nt * 16 + (lane & 15);
  int k0 = kt * 32 + ((lane >> 4) << 3) + ((p >> 2) << 4) + 2 * (p & 3);
  unsigned lo = f2bf(W[(size_t)k0 * Nc + n]);
  unsigned hi = f2bf(W[(size_t)(k0 + 1) * Nc + n]);
  P[q] = lo | (hi << 16);
}

// ---------- small kernels ----------
__global__ void k_geometry(const float* __restrict__ pos, const int* __restrict__ ei,
                           const int* __restrict__ cofs, const float* __restrict__ cell,
                           float* __restrict__ unitB, float* __restrict__ fcB,
                           float* __restrict__ rbfB, int E) {
  int e = blockIdx.x * blockDim.x + threadIdx.x;
  if (e >= E) return;
  int s = ei[e], r = ei[E + e];
  float d[3];
#pragma unroll
  for (int c = 0; c < 3; ++c) {
    float off = (float)cofs[e * 3 + 0] * cell[0 * 3 + c] +
                (float)cofs[e * 3 + 1] * cell[1 * 3 + c] +
                (float)cofs[e * 3 + 2] * cell[2 * 3 + c];
    d[c] = pos[s * 3 + c] - pos[r * 3 + c] + off;
  }
  float dist = sqrtf(d[0] * d[0] + d[1] * d[1] + d[2] * d[2]);
  float inv = __builtin_amdgcn_rcpf(dist);
#pragma unroll
  for (int c = 0; c < 3; ++c) unitB[e * 3 + c] = d[c] * inv;
  fcB[e] = (dist < CUT_F) ? 0.5f * (__cosf(PI_F * dist / CUT_F) + 1.0f) : 0.0f;
#pragma unroll
  for (int k = 0; k < NRBF; ++k)
    rbfB[e * NRBF + k] = __sinf((float)(k + 1) * (PI_F / CUT_F) * dist) * inv;
}

__global__ void k_embed(const int* __restrict__ z, const float* __restrict__ emb,
                        float* __restrict__ ns, int total) {
  int i = blockIdx.x * blockDim.x + threadIdx.x;
  if (i >= total) return;
  int n = i >> 7, h = i & (H - 1);
  ns[i] = emb[z[n] * H + h];
}

__global__ void k_accumulate(float* __restrict__ ns, const float* __restrict__ ds,
                             float* __restrict__ nv, const float* __restrict__ dv,
                             int nsz, int nvsz) {
  int i = blockIdx.x * blockDim.x + threadIdx.x;
  if (i < nsz) ns[i] += ds[i];
  else if (i < nsz + nvsz) { int j = i - nsz; nv[j] += dv[j]; }
}

__global__ void k_norm_inner_cat(const float* __restrict__ ns, const float* __restrict__ Uv,
                                 const float* __restrict__ Vv, float* __restrict__ inner,
                                 float* __restrict__ cat, int NH) {
  int i = blockIdx.x * blockDim.x + threadIdx.x;
  if (i >= NH) return;
  int n = i >> 7, h = i & (H - 1);
  float dot = 0.f, vv = 0.f;
#pragma unroll
  for (int c = 0; c < 3; ++c) {
    float u = Uv[(n * 3 + c) * H + h];
    float v = Vv[(n * 3 + c) * H + h];
    dot += u * v; vv += v * v;
  }
  inner[i] = dot;
  cat[n * (2 * H) + h]     = ns[i];
  cat[n * (2 * H) + H + h] = sqrtf(vv + 1e-12f);
}

__global__ void k_update_apply(float* __restrict__ ns, float* __restrict__ nv,
                               const float* __restrict__ a, const float* __restrict__ inner,
                               const float* __restrict__ Uv, int NH) {
  int i = blockIdx.x * blockDim.x + threadIdx.x;
  if (i >= NH) return;
  int n = i >> 7, h = i & (H - 1);
  float ass = a[n * H3 + h];
  float asv = a[n * H3 + H + h];
  float avv = a[n * H3 + 2 * H + h];
  ns[i] += ass + asv * inner[i];
#pragma unroll
  for (int c = 0; c < 3; ++c)
    nv[(n * 3 + c) * H + h] += avv * Uv[(n * 3 + c) * H + h];
}

// ---------- fused edge message kernel (16 edges / block, 4 waves) ----------
__global__ __launch_bounds__(128)
void k_message(const float* __restrict__ ns, const float* __restrict__ nv,
               const int* __restrict__ ei,
               const float* __restrict__ unitB, const float* __restrict__ fcB,
               const float* __restrict__ rbfB,
               const unsigned int* __restrict__ w1p, const float* __restrict__ b1,
               const unsigned int* __restrict__ w2p, const float* __restrict__ b2,
               const float* __restrict__ wf, const float* __restrict__ bfv,
               float* __restrict__ ds, float* __restrict__ dv, int E) {
  __shared__ __align__(32) unsigned int XsP[4 * 32 * 8];  // packed A tile (4KB)
  __shared__ __align__(32) unsigned int HsP[4 * 32 * 8];  // packed hidden  (4KB)
  __shared__ float phiS[16 * H3];                         // 24KB
  __shared__ float rbfS[16 * NRBF];
  __shared__ float unitS[16 * 3];
  __shared__ float fcS[16];
  __shared__ int   sS[16], rS[16];

  const int t = threadIdx.x;
  const int e0 = blockIdx.x * 16;

  // phase 1a: edge metadata
  if (t < 16) {
    int e = e0 + t;
    if (e < E) { sS[t] = ei[e]; rS[t] = ei[E + e]; fcS[t] = fcB[e]; }
    else       { sS[t] = -1;    rS[t] = -1;        fcS[t] = 0.f;    }
  }
  for (int i = t; i < 16 * 3; i += 128) {
    int e = e0 + i / 3;
    unitS[i] = (e < E) ? unitB[e * 3 + (i % 3)] : 0.f;
  }
  for (int i = t; i < 16 * NRBF; i += 128) {
    int e = e0 + i / NRBF;
    rbfS[i] = (e < E) ? rbfB[e * NRBF + (i % NRBF)] : 0.f;
  }
  __syncthreads();

  // phase 1b: gather ns[sender] -> packed A tile (float4 loads, dword stores)
  {
    int m  = t >> 3;
    int j0 = (t & 7) * 16;
    int s  = sS[m];
    __align__(16) float v[16];
    if (s >= 0) {
      const float4* src = (const float4*)(ns + (size_t)s * H + j0);
      ((float4*)v)[0] = src[0]; ((float4*)v)[1] = src[1];
      ((float4*)v)[2] = src[2]; ((float4*)v)[3] = src[3];
    } else {
#pragma unroll
      for (int j = 0; j < 16; ++j) v[j] = 0.f;
    }
    int kb = j0 >> 5;
    int base = j0 & 31;          // 0 or 16
#pragma unroll
    for (int d = 0; d < 8; ++d) {
      int koff = base + 2 * d;
      int lh = (koff >> 3) & 1;
      int p  = ((koff >> 4) << 2) + ((koff & 7) >> 1);
      unsigned pv = (unsigned)f2bf(v[2 * d]) | ((unsigned)f2bf(v[2 * d + 1]) << 16);
      XsP[(size_t)((kb << 5) + lh * 16 + m) * 8 + p] = pv;
    }
  }
  __syncthreads();

  const int wave = t >> 5;
  const int lane = t & 31;
  const int rowoff = (lane >> 4) << 3;
  unsigned short* Hs16 = (unsigned short*)HsP;

  // phase 2: Hs = silu(Xs @ W1 + b1)   (8 col-tiles, 2 per wave)
#pragma unroll
  for (int it = 0; it < 2; ++it) {
    int nt = wave + it * 4;
    float bv = b1[nt * 16 + (lane & 15)];
    f32x8 acc;
#pragma unroll
    for (int r = 0; r < 8; ++r) acc[r] = bv;
#pragma unroll
    for (int kb = 0; kb < 4; ++kb) {
      short16 a = load_a_packed(XsP, kb);
      short16 b = load_b_packed(w1p, H / 16, kb, nt);
      acc = wmma_bf16(a, b, acc);
    }
    // store into packed-A layout for phase 3 (k = output column)
    int k = nt * 16 + (lane & 15);
    int koff = k & 31;
    int p  = ((koff >> 4) << 2) + ((koff & 7) >> 1);
    int diBase = ((k >> 5) * 32 + ((koff >> 3) & 1) * 16) * 8 + p;
    int half = k & 1;
#pragma unroll
    for (int r = 0; r < 8; ++r)
      Hs16[(size_t)(diBase + (r + rowoff) * 8) * 2 + half] = f2bf(silu_f(acc[r]));
  }
  __syncthreads();

  // phase 3: phiS = Hs @ W2 + b2   (24 col-tiles, 6 per wave)
#pragma unroll
  for (int it = 0; it < 6; ++it) {
    int nt = wave + it * 4;
    float bv = b2[nt * 16 + (lane & 15)];
    f32x8 acc;
#pragma unroll
    for (int r = 0; r < 8; ++r) acc[r] = bv;
#pragma unroll
    for (int kb = 0; kb < 4; ++kb) {
      short16 a = load_a_packed(HsP, kb);
      short16 b = load_b_packed(w2p, H3 / 16, kb, nt);
      acc = wmma_bf16(a, b, acc);
    }
    int c = nt * 16 + (lane & 15);
#pragma unroll
    for (int r = 0; r < 8; ++r)
      phiS[(r + rowoff) * H3 + c] = acc[r];
  }
  __syncthreads();

  // phase 4: phiS *= (rbf @ wf + bf) * fc ; each thread owns 3 columns,
  // hoists its wf column into registers, reuses across all 16 edges.
#pragma unroll
  for (int jj = 0; jj < 3; ++jj) {
    int j = t + jj * 128;
    float wcol[NRBF];
#pragma unroll
    for (int k = 0; k < NRBF; ++k) wcol[k] = wf[k * H3 + j];
    float bj = bfv[j];
#pragma unroll
    for (int m = 0; m < 16; ++m) {
      float acc = bj;
#pragma unroll
      for (int k = 0; k < NRBF; ++k) acc += rbfS[m * NRBF + k] * wcol[k];
      phiS[m * H3 + j] *= acc * fcS[m];
    }
  }
  __syncthreads();

  // phase 5: gated scatter (segment_sum semantics via f32 atomics)
  for (int i = t; i < 16 * H; i += 128) {
    int m = i >> 7, h = i & (H - 1);
    int s = sS[m], r = rS[m];
    if (r < 0) continue;
    float gv = phiS[m * H3 + h];
    float ge = phiS[m * H3 + H + h];
    float ms = phiS[m * H3 + 2 * H + h];
    atomicAdd(&ds[r * H + h], ms);
#pragma unroll
    for (int c = 0; c < 3; ++c) {
      float mv = nv[(s * 3 + c) * H + h] * gv + ge * unitS[m * 3 + c];
      atomicAdd(&dv[(r * 3 + c) * H + h], mv);
    }
  }
}

// ---------- generic WMMA GEMM: C = act(A[MxK] @ Wp[KxNc packed] + bias) ----------
// block: 128 threads (4 waves) -> 16 rows x 64 cols; grid = (ceil(M/16), Nc/64)
__global__ __launch_bounds__(128)
void k_gemm_bf16(const float* __restrict__ A, const unsigned int* __restrict__ Wp,
                 const float* __restrict__ bias, float* __restrict__ C,
                 int M, int K, int Nc, int act) {
  __shared__ __align__(32) unsigned int At[8 * 32 * 8];   // K <= 256 -> 8KB
  const int t = threadIdx.x;
  const int m0 = blockIdx.x * 16;
  const int nchunks = K >> 4;                 // 8 (K=128) or 16 (K=256)
  const int cshift = (K == 256) ? 4 : 3;

  for (int ch = t; ch < 16 * nchunks; ch += 128) {
    int m  = ch >> cshift;
    int j0 = (ch & (nchunks - 1)) << 4;
    int row = m0 + m;
    __align__(16) float v[16];
    if (row < M) {
      const float4* src = (const float4*)(A + (size_t)row * K + j0);
      ((float4*)v)[0] = src[0]; ((float4*)v)[1] = src[1];
      ((float4*)v)[2] = src[2]; ((float4*)v)[3] = src[3];
    } else {
#pragma unroll
      for (int j = 0; j < 16; ++j) v[j] = 0.f;
    }
    int kb = j0 >> 5;
    int base = j0 & 31;
#pragma unroll
    for (int d = 0; d < 8; ++d) {
      int koff = base + 2 * d;
      int lh = (koff >> 3) & 1;
      int p  = ((koff >> 4) << 2) + ((koff & 7) >> 1);
      unsigned pv = (unsigned)f2bf(v[2 * d]) | ((unsigned)f2bf(v[2 * d + 1]) << 16);
      At[(size_t)((kb << 5) + lh * 16 + m) * 8 + p] = pv;
    }
  }
  __syncthreads();

  const int wave = t >> 5;
  const int lane = t & 31;
  const int nt = blockIdx.y * 4 + wave;       // 16-col tile index
  const int ntn = Nc >> 4;
  float bv = bias ? bias[nt * 16 + (lane & 15)] : 0.f;
  f32x8 acc;
#pragma unroll
  for (int r = 0; r < 8; ++r) acc[r] = bv;

  for (int kb = 0; kb < (K >> 5); ++kb) {
    short16 a = load_a_packed(At, kb);
    short16 b = load_b_packed(Wp, ntn, kb, nt);
    acc = wmma_bf16(a, b, acc);
  }

  const int rowoff = (lane >> 4) << 3;
  const int col = nt * 16 + (lane & 15);
#pragma unroll
  for (int r = 0; r < 8; ++r) {
    int row = m0 + r + rowoff;
    if (row < M) {
      float v = acc[r];
      C[(size_t)row * Nc + col] = act ? silu_f(v) : v;
    }
  }
}

// ---------- host ----------
extern "C" void kernel_launch(void* const* d_in, const int* in_sizes, int n_in,
                              void* d_out, int out_size, void* d_ws, size_t ws_size,
                              hipStream_t stream) {
  (void)n_in; (void)out_size; (void)ws_size;
  const int N = in_sizes[0];
  const int E = in_sizes[2] / 2;

  const int*   z       = (const int*)d_in[0];
  const float* pos     = (const float*)d_in[1];
  const int*   ei      = (const int*)d_in[2];
  const int*   cofs    = (const int*)d_in[3];
  const float* cell    = (const float*)d_in[4];
  const float* embed   = (const float*)d_in[5];
  const float* msg_w1  = (const float*)d_in[6];
  const float* msg_b1  = (const float*)d_in[7];
  const float* msg_w2  = (const float*)d_in[8];
  const float* msg_b2  = (const float*)d_in[9];
  const float* msg_wf  = (const float*)d_in[10];
  const float* msg_bf  = (const float*)d_in[11];
  const float* upd_u   = (const float*)d_in[12];
  const float* upd_v   = (const float*)d_in[13];
  const float* upd_w1  = (const float*)d_in[14];
  const float* upd_b1  = (const float*)d_in[15];
  const float* upd_w2  = (const float*)d_in[16];
  const float* upd_b2  = (const float*)d_in[17];
  const float* head_w1 = (const float*)d_in[18];
  const float* head_b1 = (const float*)d_in[19];
  const float* head_w2 = (const float*)d_in[20];
  const float* head_b2 = (const float*)d_in[21];

  // carve workspace (~140 MB)
  char* wp = (char*)d_ws;
  auto carve = [&](size_t bytes) -> void* {
    void* p = (void*)wp;
    wp += (bytes + 255) & ~(size_t)255;
    return p;
  };
  float* unitB  = (float*)carve((size_t)E * 3 * 4);
  float* fcB    = (float*)carve((size_t)E * 4);
  float* rbfB   = (float*)carve((size_t)E * NRBF * 4);
  float* nsB    = (float*)carve((size_t)N * H * 4);
  float* nvB    = (float*)carve((size_t)N * 3 * H * 4);
  float* dsB    = (float*)carve((size_t)N * H * 4);
  float* dvB    = (float*)carve((size_t)N * 3 * H * 4);
  float* UvB    = (float*)carve((size_t)N * 3 * H * 4);
  float* VvB    = (float*)carve((size_t)N * 3 * H * 4);
  float* innerB = (float*)carve((size_t)N * H * 4);
  float* catB   = (float*)carve((size_t)N * 2 * H * 4);
  float* aB     = (float*)carve((size_t)N * H3 * 4);
  float* hB     = (float*)carve((size_t)N * H * 4);
  // packed weights (uint32 = bf16 pair), per-matrix fragment order
  const size_t w1n  = (size_t)H * H / 2,      w2n  = (size_t)H * H3 / 2;
  const size_t uw1n = (size_t)2 * H * H / 2,  uw2n = (size_t)H * H3 / 2;
  unsigned int* w1p  = (unsigned int*)carve(NLAYER * w1n * 4);
  unsigned int* w2p  = (unsigned int*)carve(NLAYER * w2n * 4);
  unsigned int* uup  = (unsigned int*)carve(NLAYER * w1n * 4);
  unsigned int* uvp  = (unsigned int*)carve(NLAYER * w1n * 4);
  unsigned int* uw1p = (unsigned int*)carve(NLAYER * uw1n * 4);
  unsigned int* uw2p = (unsigned int*)carve(NLAYER * uw2n * 4);
  unsigned int* hw1p = (unsigned int*)carve(w1n * 4);
  unsigned int* hw2p = (unsigned int*)carve(w1n * 4);

  auto pack = [&](const float* src, unsigned int* dst, int K, int Nc) {
    int total = (K >> 1) * Nc;
    k_pack_w<<<(total + 255) / 256, 256, 0, stream>>>(src, dst, K, Nc);
  };
  for (int i = 0; i < NLAYER; ++i) {
    pack(msg_w1 + (size_t)i * H * H,      w1p  + i * w1n,  H, H);
    pack(msg_w2 + (size_t)i * H * H3,     w2p  + i * w2n,  H, H3);
    pack(upd_u  + (size_t)i * H * H,      uup  + i * w1n,  H, H);
    pack(upd_v  + (size_t)i * H * H,      uvp  + i * w1n,  H, H);
    pack(upd_w1 + (size_t)i * 2 * H * H,  uw1p + i * uw1n, 2 * H, H);
    pack(upd_w2 + (size_t)i * H * H3,     uw2p + i * uw2n, H, H3);
  }
  pack(head_w1, hw1p, H, H);
  pack(head_w2, hw2p, H, H);

  // geometry (shared across layers)
  k_geometry<<<(E + 255) / 256, 256, 0, stream>>>(pos, ei, cofs, cell, unitB, fcB, rbfB, E);

  // init node state
  k_embed<<<(N * H + 255) / 256, 256, 0, stream>>>(z, embed, nsB, N * H);
  hipMemsetAsync(nvB, 0, (size_t)N * 3 * H * 4, stream);

  for (int i = 0; i < NLAYER; ++i) {
    // message pass
    hipMemsetAsync(dsB, 0, (size_t)N * H * 4, stream);
    hipMemsetAsync(dvB, 0, (size_t)N * 3 * H * 4, stream);
    k_message<<<(E + 15) / 16, 128, 0, stream>>>(
        nsB, nvB, ei, unitB, fcB, rbfB,
        w1p + i * w1n, msg_b1 + (size_t)i * H,
        w2p + i * w2n, msg_b2 + (size_t)i * H3,
        msg_wf + (size_t)i * NRBF * H3, msg_bf + (size_t)i * H3,
        dsB, dvB, E);
    k_accumulate<<<(N * 4 * H + 255) / 256, 256, 0, stream>>>(nsB, dsB, nvB, dvB,
                                                              N * H, N * 3 * H);
    // update pass
    k_gemm_bf16<<<dim3((3 * N + 15) / 16, 2), 128, 0, stream>>>(
        nvB, uup + i * w1n, nullptr, UvB, 3 * N, H, H, 0);
    k_gemm_bf16<<<dim3((3 * N + 15) / 16, 2), 128, 0, stream>>>(
        nvB, uvp + i * w1n, nullptr, VvB, 3 * N, H, H, 0);
    k_norm_inner_cat<<<(N * H + 255) / 256, 256, 0, stream>>>(nsB, UvB, VvB, innerB, catB,
                                                              N * H);
    k_gemm_bf16<<<dim3((N + 15) / 16, 2), 128, 0, stream>>>(
        catB, uw1p + i * uw1n, upd_b1 + (size_t)i * H, hB, N, 2 * H, H, 1);
    k_gemm_bf16<<<dim3((N + 15) / 16, 6), 128, 0, stream>>>(
        hB, uw2p + i * uw2n, upd_b2 + (size_t)i * H3, aB, N, H, H3, 0);
    k_update_apply<<<(N * H + 255) / 256, 256, 0, stream>>>(nsB, nvB, aB, innerB, UvB, N * H);
  }

  // head
  k_gemm_bf16<<<dim3((N + 15) / 16, 2), 128, 0, stream>>>(nsB, hw1p, head_b1, hB, N, H, H, 1);
  k_gemm_bf16<<<dim3((N + 15) / 16, 2), 128, 0, stream>>>(hB, hw2p, head_b2, (float*)d_out,
                                                          N, H, H, 0);
}
